// KaNCDDecoder_56848187130407
// MI455X (gfx1250) — matrix operand order repaired and checked by reading, not compiled, not run on hardware
//
#include <hip/hip_runtime.h>

#define STU_NUM  100000
#define PROB_NUM 20000
#define KNOW_NUM 128
#define DIM      128
#define BATCH    8192
#define HIDDEN   512

typedef __attribute__((ext_vector_type(16))) __bf16 v16bf;
typedef __attribute__((ext_vector_type(8)))  float  v8f;
typedef unsigned int   u32;
typedef unsigned short u16;

// ---------- bf16 helpers (round-to-nearest-even), no reliance on __bf16 arith ----------
__device__ __forceinline__ __bf16 f2bf(float f) {
  u32 u = __builtin_bit_cast(u32, f);
  u32 r = u + 0x7FFFu + ((u >> 16) & 1u);
  u16 h = (u16)(r >> 16);
  return __builtin_bit_cast(__bf16, h);
}
__device__ __forceinline__ float bf2f(__bf16 b) {
  u32 u = ((u32)__builtin_bit_cast(u16, b)) << 16;
  return __builtin_bit_cast(float, u);
}
__device__ __forceinline__ float sigmoidf_(float x) {
  return 1.0f / (1.0f + __expf(-x));
}

// ---------- WMMA fragment loads, matching CDNA5 ISA 7.12.2 layouts ----------
// A (16x32 bf16): lane L, h=L>>4, M=L&15.
//   elements 0..7  = A[M][k0 + 8h + 0..7]      (VGPR0..3, K 0-7 / 8-15)
//   elements 8..15 = A[M][k0 + 16 + 8h + 0..7] (VGPR4..7, K 16-23 / 24-31)
__device__ __forceinline__ v16bf load_a_frag(const __bf16* __restrict__ A, int lda,
                                             int m0, int k0, int lane) {
  int m = m0 + (lane & 15);
  int h = lane >> 4;
  const __bf16* p = A + (size_t)m * lda + k0 + 8 * h;
  union { v16bf v; uint4 q[2]; } u;
  u.q[0] = *(const uint4*)(p);
  u.q[1] = *(const uint4*)(p + 16);
  return u.v;
}
// B (32x16 bf16), pre-packed column-major Bp[n][k] (k contiguous):
//   lane L, h=L>>4, N=L&15: elements 0..7 = Bp[N][k0+8h+0..7], 8..15 = Bp[N][k0+16+8h+0..7]
__device__ __forceinline__ v16bf load_b_frag(const __bf16* __restrict__ Bp, int ldb,
                                             int n0, int k0, int lane) {
  int n = n0 + (lane & 15);
  int h = lane >> 4;
  const __bf16* p = Bp + (size_t)n * ldb + k0 + 8 * h;
  union { v16bf v; uint4 q[2]; } u;
  u.q[0] = *(const uint4*)(p);
  u.q[1] = *(const uint4*)(p + 16);
  return u.v;
}
__device__ __forceinline__ v8f wmma_bf16(v16bf a, v16bf b, v8f c) {
  return __builtin_amdgcn_wmma_f32_16x16x32_bf16(false, a, false, b, (short)0, c,
                                                 false, false);
}

// ---------- pack kernels ----------
// Gather embeddings, scale by w_stat / w_kdiff, convert to bf16.
__global__ void k_prep(const float* __restrict__ z, const int* __restrict__ sid,
                       const int* __restrict__ eid, const float* __restrict__ w_stat,
                       const float* __restrict__ w_kdiff,
                       __bf16* __restrict__ As, __bf16* __restrict__ Ae) {
  int idx = blockIdx.x * blockDim.x + threadIdx.x;   // [0, BATCH*DIM)
  int b = idx >> 7;
  int d = idx & (DIM - 1);
  float zs = z[(size_t)sid[b] * DIM + d];
  float ze = z[(size_t)eid[b] * DIM + d];
  As[idx] = f2bf(zs * w_stat[d]);
  Ae[idx] = f2bf(ze * w_kdiff[d]);
}

// knowledge_ts rows == columns of B for A·K^T, so a straight convert gives Bp[n][d].
__global__ void k_pack_kt(const float* __restrict__ z, __bf16* __restrict__ Bkt) {
  int idx = blockIdx.x * blockDim.x + threadIdx.x;   // [0, KNOW_NUM*DIM)
  Bkt[idx] = f2bf(z[(size_t)(STU_NUM + PROB_NUM) * DIM + idx]);
}

// Transpose-pack W (KxN row-major f32) -> Wp[n][k] bf16.
__global__ void k_pack_w(const float* __restrict__ W, __bf16* __restrict__ Wp,
                         int K, int N) {
  int idx = blockIdx.x * blockDim.x + threadIdx.x;
  if (idx >= K * N) return;
  int n = idx / K;
  int k = idx - n * K;
  Wp[idx] = f2bf(W[(size_t)k * N + n]);
}

// ---------- stage 1: two shared-B GEMMs + sigmoid diff + knowledge mask ----------
// One wave computes a 32x32 tile of both stat_emb and k_difficulty.
__global__ void __launch_bounds__(32) k_stage1(
    const __bf16* __restrict__ As, const __bf16* __restrict__ Ae,
    const __bf16* __restrict__ Bkt, const float* __restrict__ kp,
    const float* __restrict__ b_stat, const float* __restrict__ b_kdiff,
    __bf16* __restrict__ state) {
  int lane = threadIdx.x;
  int m0 = blockIdx.x * 32;
  int n0 = blockIdx.y * 32;

  v8f z0 = {};
  v8f s00 = z0, s01 = z0, s10 = z0, s11 = z0;
  v8f d00 = z0, d01 = z0, d10 = z0, d11 = z0;

  for (int kk = 0; kk < DIM; kk += 32) {
    v16bf a0 = load_a_frag(As, DIM, m0,      kk, lane);
    v16bf a1 = load_a_frag(As, DIM, m0 + 16, kk, lane);
    v16bf e0 = load_a_frag(Ae, DIM, m0,      kk, lane);
    v16bf e1 = load_a_frag(Ae, DIM, m0 + 16, kk, lane);
    v16bf b0 = load_b_frag(Bkt, DIM, n0,      kk, lane);
    v16bf b1 = load_b_frag(Bkt, DIM, n0 + 16, kk, lane);
    s00 = wmma_bf16(a0, b0, s00); s01 = wmma_bf16(a0, b1, s01);
    s10 = wmma_bf16(a1, b0, s10); s11 = wmma_bf16(a1, b1, s11);
    d00 = wmma_bf16(e0, b0, d00); d01 = wmma_bf16(e0, b1, d01);
    d10 = wmma_bf16(e1, b0, d10); d11 = wmma_bf16(e1, b1, d11);
  }

  float bs = b_stat[0];
  float bk = b_kdiff[0];
  int h  = lane >> 4;
  int nb = lane & 15;

  v8f S[2][2] = {{s00, s01}, {s10, s11}};
  v8f D[2][2] = {{d00, d01}, {d10, d11}};
  #pragma unroll
  for (int i = 0; i < 2; ++i) {
    #pragma unroll
    for (int j = 0; j < 2; ++j) {
      int n = n0 + 16 * j + nb;
      #pragma unroll
      for (int g = 0; g < 8; ++g) {
        int m = m0 + 16 * i + g + 8 * h;
        float sv = sigmoidf_(S[i][j][g] + bs);
        float dv = sigmoidf_(D[i][j][g] + bk);
        state[(size_t)m * KNOW_NUM + n] =
            f2bf(kp[(size_t)m * KNOW_NUM + n] * (sv - dv));
      }
    }
  }
}

// ---------- generic WMMA GEMM + bias + tanh (bf16 in/out, f32 accumulate) ----------
__global__ void __launch_bounds__(32) k_gemm_tanh(
    const __bf16* __restrict__ A, const __bf16* __restrict__ Bp,
    const float* __restrict__ bias, __bf16* __restrict__ out, int N, int K) {
  int lane = threadIdx.x;
  int m0 = blockIdx.x * 32;
  int n0 = blockIdx.y * 32;

  v8f z0 = {};
  v8f c00 = z0, c01 = z0, c10 = z0, c11 = z0;

  for (int kk = 0; kk < K; kk += 32) {
    v16bf a0 = load_a_frag(A, K, m0,      kk, lane);
    v16bf a1 = load_a_frag(A, K, m0 + 16, kk, lane);
    v16bf b0 = load_b_frag(Bp, K, n0,      kk, lane);
    v16bf b1 = load_b_frag(Bp, K, n0 + 16, kk, lane);
    c00 = wmma_bf16(a0, b0, c00); c01 = wmma_bf16(a0, b1, c01);
    c10 = wmma_bf16(a1, b0, c10); c11 = wmma_bf16(a1, b1, c11);
  }

  int h  = lane >> 4;
  int nb = lane & 15;
  v8f C[2][2] = {{c00, c01}, {c10, c11}};
  #pragma unroll
  for (int i = 0; i < 2; ++i) {
    #pragma unroll
    for (int j = 0; j < 2; ++j) {
      int n = n0 + 16 * j + nb;
      float bn = bias[n];
      #pragma unroll
      for (int g = 0; g < 8; ++g) {
        int m = m0 + 16 * i + g + 8 * h;
        out[(size_t)m * N + n] = f2bf(tanhf(C[i][j][g] + bn));
      }
    }
  }
}

// ---------- head: sigmoid(h3 @ W4 + b4), N=1 matrix-vector ----------
__global__ void k_head(const __bf16* __restrict__ h3, const float* __restrict__ W4,
                       const float* __restrict__ b4, float* __restrict__ out) {
  int b = blockIdx.x * blockDim.x + threadIdx.x;
  if (b >= BATCH) return;
  const __bf16* row = h3 + (size_t)b * (HIDDEN / 4);
  float acc = 0.0f;
  #pragma unroll 8
  for (int d = 0; d < HIDDEN / 4; ++d) acc += bf2f(row[d]) * W4[d];
  out[b] = sigmoidf_(acc + b4[0]);
}

extern "C" void kernel_launch(void* const* d_in, const int* in_sizes, int n_in,
                              void* d_out, int out_size, void* d_ws, size_t ws_size,
                              hipStream_t stream) {
  const float* z       = (const float*)d_in[0];
  const int*   sid     = (const int*)  d_in[1];
  const int*   eid     = (const int*)  d_in[2];
  const float* kp      = (const float*)d_in[3];
  const float* w_stat  = (const float*)d_in[4];
  const float* b_stat  = (const float*)d_in[5];
  const float* w_kdiff = (const float*)d_in[6];
  const float* b_kdiff = (const float*)d_in[7];
  const float* W1      = (const float*)d_in[8];
  const float* b1      = (const float*)d_in[9];
  const float* W2      = (const float*)d_in[10];
  const float* b2      = (const float*)d_in[11];
  const float* W3      = (const float*)d_in[12];
  const float* b3      = (const float*)d_in[13];
  const float* W4      = (const float*)d_in[14];
  const float* b4      = (const float*)d_in[15];
  float* out = (float*)d_out;

  char* ws = (char*)d_ws;
  size_t off = 0;
  auto wsalloc = [&](size_t bytes) {
    char* p = ws + off;
    off += (bytes + 255) & ~(size_t)255;
    return p;
  };
  __bf16* As    = (__bf16*)wsalloc((size_t)BATCH * DIM * 2);
  __bf16* Ae    = (__bf16*)wsalloc((size_t)BATCH * DIM * 2);
  __bf16* Bkt   = (__bf16*)wsalloc((size_t)KNOW_NUM * DIM * 2);
  __bf16* W1p   = (__bf16*)wsalloc((size_t)KNOW_NUM * HIDDEN * 2);
  __bf16* W2p   = (__bf16*)wsalloc((size_t)HIDDEN * (HIDDEN / 2) * 2);
  __bf16* W3p   = (__bf16*)wsalloc((size_t)(HIDDEN / 2) * (HIDDEN / 4) * 2);
  __bf16* state = (__bf16*)wsalloc((size_t)BATCH * KNOW_NUM * 2);
  __bf16* h1    = (__bf16*)wsalloc((size_t)BATCH * HIDDEN * 2);
  __bf16* h2    = (__bf16*)wsalloc((size_t)BATCH * (HIDDEN / 2) * 2);
  __bf16* h3    = (__bf16*)wsalloc((size_t)BATCH * (HIDDEN / 4) * 2);
  (void)ws_size; (void)in_sizes; (void)n_in; (void)out_size;

  // 1) pack operands to bf16 / WMMA-friendly layouts
  k_prep<<<(BATCH * DIM) / 256, 256, 0, stream>>>(z, sid, eid, w_stat, w_kdiff, As, Ae);
  k_pack_kt<<<(KNOW_NUM * DIM) / 256, 256, 0, stream>>>(z, Bkt);
  k_pack_w<<<(KNOW_NUM * HIDDEN + 255) / 256, 256, 0, stream>>>(W1, W1p, KNOW_NUM, HIDDEN);
  k_pack_w<<<(HIDDEN * (HIDDEN / 2) + 255) / 256, 256, 0, stream>>>(W2, W2p, HIDDEN, HIDDEN / 2);
  k_pack_w<<<((HIDDEN / 2) * (HIDDEN / 4) + 255) / 256, 256, 0, stream>>>(W3, W3p, HIDDEN / 2, HIDDEN / 4);

  // 2) fused dual GEMM + sigmoid diff + knowledge mask -> state (8192x128 bf16)
  k_stage1<<<dim3(BATCH / 32, KNOW_NUM / 32), 32, 0, stream>>>(
      As, Ae, Bkt, kp, b_stat, b_kdiff, state);

  // 3) MLP: tanh(state@W1+b1) -> tanh(@W2+b2) -> tanh(@W3+b3)
  k_gemm_tanh<<<dim3(BATCH / 32, HIDDEN / 32), 32, 0, stream>>>(
      state, W1p, b1, h1, HIDDEN, KNOW_NUM);
  k_gemm_tanh<<<dim3(BATCH / 32, (HIDDEN / 2) / 32), 32, 0, stream>>>(
      h1, W2p, b2, h2, HIDDEN / 2, HIDDEN);
  k_gemm_tanh<<<dim3(BATCH / 32, (HIDDEN / 4) / 32), 32, 0, stream>>>(
      h2, W3p, b3, h3, HIDDEN / 4, HIDDEN / 2);

  // 4) sigmoid head (N=1)
  k_head<<<BATCH / 256, 256, 0, stream>>>(h3, W4, b4, out);
}
